// MergingModel_30374008717888
// MI455X (gfx1250) — matrix operand order repaired and checked by reading, not compiled, not run on hardware
//
#include <hip/hip_runtime.h>
#include <hip/hip_bf16.h>

// ---------------- problem constants (match reference) ----------------
#define N_PRED   2000000
#define N_OBS    1000000
#define RASU     200000
#define N_IMAGES 2000
#define HMAX     60
#define GRID_SZ  121          // 2*HMAX+1
#define MC       32
#define D_META   5
#define LOG2PI   1.8378770664093453f
#define KL_W     1.0f

// workspace layout (bytes)
#define ZT_BYTES   ((size_t)RASU * MC * 4)     // 25,600,000
#define QS_BYTES   ((size_t)RASU * 4)          //    800,000
// acc: 3*N_IMAGES floats, then 1 float klsum

typedef __attribute__((ext_vector_type(16))) _Float16 v16h;
typedef __attribute__((ext_vector_type(8)))  float    v8f;
typedef __attribute__((address_space(3)))    float    lds_f32;

#define ZPAD 36   // LDS row stride (floats) for zT staging: 144B kills 256B bank periodicity

// 3 reindexing ops, row-major: _h[i] = M[i][0]*h0 + M[i][1]*h1 + M[i][2]*h2
__constant__ int OPMAT[27] = {
     1, 0, 0,   0, 1, 0,   0, 0, 1,    // identity
    -1, 0, 0,   0,-1, 0,   0, 0,-1,    // inversion
     0, 1, 0,   1, 0, 0,   0, 0,-1     // permutation op
};

// ---------------- kernel 0: zero accumulators ----------------
__global__ void k_zero(float* p, int n) {
    for (int i = threadIdx.x; i < n; i += blockDim.x) p[i] = 0.0f;
}

// ---------------- kernel 1: zT rsample + qscale + KL reduction ----------------
__global__ void k_zt_kl(const float* __restrict__ q_loc,
                        const float* __restrict__ q_raw,
                        const float* __restrict__ eps,
                        float* __restrict__ zT,
                        float* __restrict__ qscale,
                        float* __restrict__ klsum) {
    __shared__ float sred[256];
    int r = blockIdx.x * blockDim.x + threadIdx.x;
    float kl = 0.0f;
    if (r < RASU) {
        float x  = q_raw[r];
        float qs = (x > 20.0f) ? x : log1pf(expf(x));   // softplus
        qs += 1e-6f;
        float ql = q_loc[r];
        kl = -logf(qs) + 0.5f * (qs * qs + ql * ql) - 0.5f;
        qscale[r] = qs;
        float4* zrow = (float4*)(zT + (size_t)r * MC);  // 128B row, 16B aligned
        #pragma unroll
        for (int mq = 0; mq < MC / 4; ++mq) {
            float4 v;
            v.x = fmaf(qs, eps[(size_t)(4 * mq + 0) * RASU + r], ql);
            v.y = fmaf(qs, eps[(size_t)(4 * mq + 1) * RASU + r], ql);
            v.z = fmaf(qs, eps[(size_t)(4 * mq + 2) * RASU + r], ql);
            v.w = fmaf(qs, eps[(size_t)(4 * mq + 3) * RASU + r], ql);
            zrow[mq] = v;
        }
    }
    sred[threadIdx.x] = kl;
    __syncthreads();
    for (int s = 128; s > 0; s >>= 1) {
        if (threadIdx.x < s) sred[threadIdx.x] += sred[threadIdx.x + s];
        __syncthreads();
    }
    if (threadIdx.x == 0) atomicAdd(klsum, sred[0]);
}

// ---------------- kernel 2: main fused per-op pass ----------------
// block = 256 threads = 8 waves; each wave handles 32 predictions = 16 observations.
// grid.x covers N_PRED/256, grid.y = op index.
__global__ __launch_bounds__(256) void k_main(
        const int*   __restrict__ hkl,
        const float* __restrict__ I,
        const float* __restrict__ SigI,
        const int*   __restrict__ image_id,
        const float* __restrict__ metadata,
        const int*   __restrict__ asu_table,
        const float* __restrict__ q_loc,
        const float* __restrict__ qscale,
        const float* __restrict__ zT,
        const float* __restrict__ scale_w,
        const float* __restrict__ scale_b,
        const float* __restrict__ image_bias,
        float*       __restrict__ acc) {

    __shared__ _Float16 sm_feat[8][32][16];    // [wave][local pred][K padded] (32B rows)
    __shared__ float    sm_z   [8][32][ZPAD];  // async-staged zT rows, padded stride
    __shared__ int      sm_img [8][32];
    __shared__ float    sm_dot [8][32];
    __shared__ float    sm_scl [8][32];
    __shared__ float    sm_ps  [8][32];

    const int tid  = threadIdx.x;
    const int lane = tid & 31;
    const int wave = tid >> 5;
    const int op   = blockIdx.y;
    const int wbase = blockIdx.x * 256 + wave * 32;   // first prediction of wave
    const int p     = wbase + lane;

    // ---- gather features for this lane's prediction ----
    float f[9];
    #pragma unroll
    for (int k = 0; k < 9; ++k) f[k] = 0.0f;
    int refl = 0, img = 0;
    if (p < N_PRED) {
        const int h0 = hkl[3 * p + 0];
        const int h1 = hkl[3 * p + 1];
        const int h2 = hkl[3 * p + 2];
        const int* M = OPMAT + op * 9;
        const int a0 = M[0] * h0 + M[1] * h1 + M[2] * h2;
        const int a1 = M[3] * h0 + M[4] * h1 + M[5] * h2;
        const int a2 = M[6] * h0 + M[7] * h1 + M[8] * h2;
        const int flat = ((a0 + HMAX) * GRID_SZ + (a1 + HMAX)) * GRID_SZ + (a2 + HMAX);
        refl = asu_table[flat];
        const int o = p >> 1;                 // harmonic_id = p // 2
        f[0] = q_loc[refl];
        f[1] = qscale[refl];
        f[2] = I[o];
        f[3] = SigI[o];
        #pragma unroll
        for (int k = 0; k < D_META; ++k) f[4 + k] = metadata[(size_t)p * D_META + k];
        img = image_id[p];
    }

    // ---- async-stage this lane's 128B zT row into LDS (ASYNCcnt-tracked),
    //      overlapped with feature staging + WMMA + expf below ----
    {
        lds_f32* ldsp = (lds_f32*)&sm_z[wave][lane][0];
        const unsigned long long gaddr =
            (unsigned long long)(uintptr_t)(zT + (size_t)refl * MC);
        // INST_OFFSET applies to both LDS and global addresses (ISA §10 async)
        #define ZT_ASYNC(OFF)                                                  \
            asm volatile("global_load_async_to_lds_b128 %0, %1, off offset:" OFF \
                         :: "v"(ldsp), "v"(gaddr) : "memory")
        ZT_ASYNC("0");  ZT_ASYNC("16"); ZT_ASYNC("32"); ZT_ASYNC("48");
        ZT_ASYNC("64"); ZT_ASYNC("80"); ZT_ASYNC("96"); ZT_ASYNC("112");
        #undef ZT_ASYNC
    }

    #pragma unroll
    for (int k = 0; k < 9; ++k)  sm_feat[wave][lane][k] = (_Float16)f[k];
    #pragma unroll
    for (int k = 9; k < 16; ++k) sm_feat[wave][lane][k] = (_Float16)0.0f;
    sm_img[wave][lane] = img;
    __syncthreads();

    // ---- B operand: scale_w in column N=0, zero elsewhere ----
    // 16-bit B 32x16 layout: VGPR r<4, lanes 0-15 hold K=2r,2r+1 at N=lane;
// lanes 16-31 hold K=8+2r,9+2r at N=lane-16.  Only lanes 0 (K=0..7) and 16 (K=8) nonzero.
    v16h b;
    #pragma unroll
    for (int i = 0; i < 16; ++i) b[i] = (_Float16)0.0f;
    if (lane == 0) {
        #pragma unroll
        for (int k = 0; k < 8; ++k) b[k] = (_Float16)scale_w[k];
    }
    if (lane == 16) b[0] = (_Float16)scale_w[8];

    // ---- two WMMAs cover the wave's 32 predictions (16 rows each) ----
    #pragma unroll
    for (int g = 0; g < 2; ++g) {
        // 16-bit A 16x32 layout: lanes 0-15 carry K=0..7 in v0-3; lanes 16-31 carry
        // K=8..15.  Row halves 8..15 in LDS are {feat8, 0...}, so BOTH lane groups
        // load 16B from the same row (offset 0 or 16) -- branchless address select.
        const int row = g * 16 + (lane & 15);
        const _Float16* rowp = &sm_feat[wave][row][(lane < 16) ? 0 : 8];
        v16h a;
        #pragma unroll
        for (int i = 0; i < 8; ++i)  a[i] = rowp[i];
        #pragma unroll
        for (int i = 8; i < 16; ++i) a[i] = (_Float16)0.0f;
        v8f c;
        #pragma unroll
        for (int i = 0; i < 8; ++i) c[i] = 0.0f;
        v8f d = __builtin_amdgcn_wmma_f32_16x16x32_f16(
            false, a, false, b, (short)0, c, false, false);
        // D layout: VGPR r, lane 0 -> (M=r, N=0); lane 16 -> (M=8+r, N=0)
        if (lane == 0) {
            #pragma unroll
            for (int r2 = 0; r2 < 8; ++r2) sm_dot[wave][g * 16 + r2] = d[r2];
        }
        if (lane == 16) {
            #pragma unroll
            for (int r2 = 0; r2 < 8; ++r2) sm_dot[wave][g * 16 + 8 + r2] = d[r2];
        }
    }
    __syncthreads();

    // ---- per-prediction scale = exp(logit + b + image_bias) ----
    float sc = 0.0f;
    if (p < N_PRED) sc = expf(sm_dot[wave][lane] + scale_b[0] + image_bias[img]);
    sm_scl[wave][lane] = sc;
    __syncthreads();

    // ---- drain the async zT staging before reading it back ----
    asm volatile("s_wait_asynccnt 0" ::: "memory");

    // ---- observation phase: lane pair (l, l+16) splits the MC dimension ----
    const int oloc = lane & 15;
    const int mh   = lane >> 4;                       // 0: m=0..15, 1: m=16..31
    const int ob   = (wbase >> 1) + oloc;
    const float s0 = sm_scl[wave][2 * oloc + 0];
    const float s1 = sm_scl[wave][2 * oloc + 1];
    float sumsq = 0.0f;
    if (ob < N_OBS) {
        const float Io = I[ob];
        const float4* z0 = (const float4*)&sm_z[wave][2 * oloc + 0][mh * 16];
        const float4* z1 = (const float4*)&sm_z[wave][2 * oloc + 1][mh * 16];
        #pragma unroll
        for (int q = 0; q < 4; ++q) {
            const float4 za = z0[q];
            const float4 zb = z1[q];
            float d0 = fmaf(s0, za.x, s1 * zb.x) - Io;
            float d1 = fmaf(s0, za.y, s1 * zb.y) - Io;
            float d2 = fmaf(s0, za.z, s1 * zb.z) - Io;
            float d3 = fmaf(s0, za.w, s1 * zb.w) - Io;
            sumsq += d0 * d0 + d1 * d1 + d2 * d2 + d3 * d3;
        }
    }
    sm_ps[wave][lane] = sumsq;
    __syncthreads();

    if (lane < 16 && ob < N_OBS) {
        const float Sig = SigI[ob];
        const float tot = sm_ps[wave][lane] + sm_ps[wave][lane + 16];
        // sum over m of ll[ob, m]
        const float contrib = -0.5f * tot / (Sig * Sig)
                              - (float)MC * (logf(Sig) + 0.5f * LOG2PI);
        const int imgo = sm_img[wave][2 * oloc + 1];   // scatter .set: last write wins
        atomicAdd(&acc[op * N_IMAGES + imgo], contrib);
    }
}

// ---------------- kernel 3: finalize (argmax over ops, ELBO) ----------------
__global__ void k_final(const float* __restrict__ acc,
                        const float* __restrict__ klsum,
                        float* __restrict__ out) {
    __shared__ float sred[256];
    const int t = threadIdx.x;
    const float inv = 1.0f / (float)(MC * MC);       // segment_sum/MC then mean over MC
    float local = 0.0f;
    for (int i = t; i < N_IMAGES; i += 256) {
        const float a0 = acc[0 * N_IMAGES + i] * inv;
        const float a1 = acc[1 * N_IMAGES + i] * inv;
        const float a2 = acc[2 * N_IMAGES + i] * inv;
        float best = a0; int idx = 0;
        if (a1 > best) { best = a1; idx = 1; }
        if (a2 > best) { best = a2; idx = 2; }
        out[1 + i] = (float)idx;
        local += best;
    }
    sred[t] = local;
    __syncthreads();
    for (int s = 128; s > 0; s >>= 1) {
        if (t < s) sred[t] += sred[t + s];
        __syncthreads();
    }
    if (t == 0) {
        const float kl = klsum[0] / (float)RASU;
        out[0] = -(sred[0] / (float)N_IMAGES) + KL_W * kl;
    }
}

// ---------------- launcher ----------------
extern "C" void kernel_launch(void* const* d_in, const int* in_sizes, int n_in,
                              void* d_out, int out_size, void* d_ws, size_t ws_size,
                              hipStream_t stream) {
    const int*   hkl        = (const int*)  d_in[0];
    const float* I          = (const float*)d_in[1];
    const float* SigI       = (const float*)d_in[2];
    const int*   image_id   = (const int*)  d_in[3];
    const float* metadata   = (const float*)d_in[4];
    // d_in[5] = harmonic_id (== p>>1, computed inline)
    const int*   asu_table  = (const int*)  d_in[6];
    const float* q_loc      = (const float*)d_in[7];
    const float* q_raw      = (const float*)d_in[8];
    const float* eps        = (const float*)d_in[9];
    const float* scale_w    = (const float*)d_in[10];
    const float* scale_b    = (const float*)d_in[11];
    const float* image_bias = (const float*)d_in[12];
    float* out = (float*)d_out;

    char* ws = (char*)d_ws;
    float* zT     = (float*)ws;
    float* qscale = (float*)(ws + ZT_BYTES);
    float* acc    = (float*)(ws + ZT_BYTES + QS_BYTES);
    float* klsum  = acc + 3 * N_IMAGES;

    k_zero<<<1, 256, 0, stream>>>(acc, 3 * N_IMAGES + 1);

    k_zt_kl<<<(RASU + 255) / 256, 256, 0, stream>>>(q_loc, q_raw, eps, zT, qscale, klsum);

    dim3 grid((N_PRED + 255) / 256, 3);
    k_main<<<grid, 256, 0, stream>>>(hkl, I, SigI, image_id, metadata, asu_table,
                                     q_loc, qscale, zT, scale_w, scale_b, image_bias,
                                     acc);

    k_final<<<1, 256, 0, stream>>>(acc, klsum, out);
}
